// ImprovedGCN_69492570849587
// MI455X (gfx1250) — compile-verified, matrix operand order checked by reading, not run on hardware
//
#include <hip/hip_runtime.h>
#include <hip/hip_bf16.h>
#include <stdint.h>

// ---------------------------------------------------------------------------
// Heterogeneous GraphSAGE (author/paper) for MI455X (gfx1250).
// GEMMs in bf16 via v_wmma_f32_16x16x32_bf16 (f32 accumulate), weights
// pre-transposed so A and B tiles stage via b128; async global->LDS double
// buffering via GLOBAL_LOAD_ASYNC_TO_LDS_B128 (ASYNCcnt-tracked).
// Aggregation via f32 global atomics, LN/softmax via wave32 shuffles.
// ---------------------------------------------------------------------------

#define NA 50000
#define NP 100000
#define DIN 128
#define HD 256
#define NC 40
#define NLAYERS 2

#if defined(__has_builtin)
#if __has_builtin(__builtin_amdgcn_global_load_async_to_lds_b128) && \
    __has_builtin(__builtin_amdgcn_s_wait_asynccnt)
#define USE_ASYNC_LDS 1
#endif
#endif

typedef __attribute__((ext_vector_type(16))) __bf16 v16bf;
typedef __attribute__((ext_vector_type(8)))  float  v8f;
typedef __attribute__((ext_vector_type(4)))  int    v4i;

// builtin expects (AS1 v4i* src, AS3 v4i* dst, imm offset, imm cpol)
#define AS1_V4I(p) ((__attribute__((address_space(1))) v4i*)                 \
                    (__attribute__((address_space(1))) void*)(void*)(p))
#define AS3_V4I(p) ((__attribute__((address_space(3))) v4i*)                 \
                    (__attribute__((address_space(3))) void*)(void*)(p))

__device__ __forceinline__ unsigned short f2bf(float f) {
    union { float f; unsigned u; } x; x.f = f;
    unsigned r = x.u + 0x7FFFu + ((x.u >> 16) & 1u);   // round-to-nearest-even
    return (unsigned short)(r >> 16);
}

// ------------------------------ elementwise --------------------------------

__global__ void cast_bf16_kernel(const float* __restrict__ in,
                                 unsigned short* __restrict__ out, int n) {
    int i = blockIdx.x * blockDim.x + threadIdx.x;
    if (i < n) out[i] = f2bf(in[i]);
}

// fp32 [K][N] -> bf16 [N][K] (weight pre-transpose for the GEMM B operand)
__global__ void cast_bf16_t_kernel(const float* __restrict__ in,
                                   unsigned short* __restrict__ out, int N, int K) {
    int i = blockIdx.x * blockDim.x + threadIdx.x;
    if (i >= N * K) return;
    int n = i / K, k = i - n * K;
    out[i] = f2bf(in[(long long)k * N + n]);
}

__global__ void fill_zero_kernel(float* __restrict__ p, long long n) {
    long long i = (long long)blockIdx.x * blockDim.x + threadIdx.x;
    if (i < n) p[i] = 0.0f;
}

// ------------------------------ aggregation --------------------------------

__global__ void scatter_mean_kernel(const float* __restrict__ hsrc,
                                    const int* __restrict__ src,
                                    const int* __restrict__ dst,
                                    float* __restrict__ acc,
                                    float* __restrict__ deg, int nE) {
    long long t = (long long)blockIdx.x * blockDim.x + threadIdx.x;
    const int chunks = HD / 4;                // 64
    long long e = t / chunks;
    int c = (int)(t % chunks);
    if (e >= nE) return;
    int s = src[e], d = dst[e];
    const float4 v = *(const float4*)(hsrc + (long long)s * HD + c * 4);
    float* a = acc + (long long)d * HD + c * 4;
    atomicAdd(a + 0, v.x);
    atomicAdd(a + 1, v.y);
    atomicAdd(a + 2, v.z);
    atomicAdd(a + 3, v.w);
    if (c == 0) atomicAdd(deg + d, 1.0f);
}

__global__ void mean_cast_kernel(const float* __restrict__ acc,
                                 const float* __restrict__ deg,
                                 unsigned short* __restrict__ out_bf, long long n) {
    long long i = (long long)blockIdx.x * blockDim.x + threadIdx.x;
    if (i >= n) return;
    int row = (int)(i / HD);
    float m = acc[i] / fmaxf(deg[row], 1.0f);
    out_bf[i] = f2bf(m);
}

// --------------------------------- GEMM ------------------------------------
// C[M,N] (f32) = A[M,K](bf16, row major) @ Bt[N,K](bf16, pre-transposed)
// (+bias) (+=C if acc) (relu opt). 256 threads = 8 waves; block tile 128x64,
// wave tile 32x32 (2x2 of v_wmma_f32_16x16x32_bf16), K-step 32.

__global__ __launch_bounds__(256) void gemm_bf16_kernel(
        const unsigned short* __restrict__ Ag,
        const unsigned short* __restrict__ Bt,
        const float* __restrict__ bias,
        float* __restrict__ Cg,
        unsigned short* __restrict__ Cbf,
        int M, int N, int K, int accumulate, int relu) {
    __shared__ __align__(16) unsigned short lA[2][128 * 32];   // [m][k]
    __shared__ __align__(16) unsigned short lB[2][64 * 32];    // [n][k]

    const int tid  = threadIdx.x;
    const int lane = tid & 31;
    const int wave = tid >> 5;
    const int wm   = wave & 3;            // 0..3 -> 32-row band
    const int wn   = wave >> 2;           // 0..1 -> 32-col band
    const int half = lane >> 4;
    const int l16  = lane & 15;
    const int bm0  = blockIdx.y * 128;
    const int bn0  = blockIdx.x * 64;

    // per-thread tile staging coordinates
    const int ar = tid >> 1, ac = (tid & 1) * 16;   // A: 2 x b128 per thread
    const int bn = tid >> 2, bk = (tid & 3) * 8;    // B: 1 x b128 per thread

    v8f acc[2][2];
    for (int mi = 0; mi < 2; ++mi)
        for (int ni = 0; ni < 2; ++ni)
            acc[mi][ni] = (v8f){0.f, 0.f, 0.f, 0.f, 0.f, 0.f, 0.f, 0.f};

    union Frag { v16bf v; uint4 u[2]; };

    // fragment K layout (16-bit WMMA): lanes 0-15 hold K 0-7 & 16-23,
    // lanes 16-31 hold K 8-15 & 24-31 -> two contiguous 16B LDS loads.
    auto compute = [&](const unsigned short* la, const unsigned short* lb) {
        Frag fa[2], fb[2];
        for (int mi = 0; mi < 2; ++mi) {
            int r = wm * 32 + mi * 16 + l16;
            fa[mi].u[0] = *(const uint4*)(la + r * 32 + half * 8);
            fa[mi].u[1] = *(const uint4*)(la + r * 32 + 16 + half * 8);
        }
        for (int ni = 0; ni < 2; ++ni) {
            int n = wn * 32 + ni * 16 + l16;
            fb[ni].u[0] = *(const uint4*)(lb + n * 32 + half * 8);
            fb[ni].u[1] = *(const uint4*)(lb + n * 32 + 16 + half * 8);
        }
        for (int mi = 0; mi < 2; ++mi)
            for (int ni = 0; ni < 2; ++ni)
                acc[mi][ni] = __builtin_amdgcn_wmma_f32_16x16x32_bf16(
                        false, fa[mi].v, false, fb[ni].v,
                        (short)0, acc[mi][ni], false, false);
    };

    const bool interior = (bm0 + 128 <= M) && (bn0 + 64 <= N);

#if defined(USE_ASYNC_LDS)
    if (interior) {
        // async double-buffered pipeline: 3 async b128 per thread per tile
        auto issue = [&](int kt, int buf) {
            const unsigned short* ga = Ag + (size_t)(bm0 + ar) * K + kt + ac;
            __builtin_amdgcn_global_load_async_to_lds_b128(
                AS1_V4I(ga),     AS3_V4I(&lA[buf][ar * 32 + ac]),     0, 0);
            __builtin_amdgcn_global_load_async_to_lds_b128(
                AS1_V4I(ga + 8), AS3_V4I(&lA[buf][ar * 32 + ac + 8]), 0, 0);
            const unsigned short* gb = Bt + (size_t)(bn0 + bn) * K + kt + bk;
            __builtin_amdgcn_global_load_async_to_lds_b128(
                AS1_V4I(gb),     AS3_V4I(&lB[buf][bn * 32 + bk]),     0, 0);
        };
        const int nkt = K >> 5;
        issue(0, 0);
        for (int t = 0; t < nkt; ++t) {
            int cur = t & 1;
            if (t + 1 < nkt) {
                issue((t + 1) << 5, cur ^ 1);
                __builtin_amdgcn_s_wait_asynccnt(3);   // current tile's 3 ops done
            } else {
                __builtin_amdgcn_s_wait_asynccnt(0);
            }
            __syncthreads();
            compute(&lA[cur][0], &lB[cur][0]);
            __syncthreads();                           // protect buf before reuse
        }
    } else
#endif
    {
        // synchronous path with zero-fill guards (boundary blocks / fallback)
        for (int kt = 0; kt < K; kt += 32) {
            {
                uint4 a0 = {0, 0, 0, 0}, a1 = {0, 0, 0, 0};
                int gr = bm0 + ar;
                if (gr < M) {
                    const uint4* p = (const uint4*)(Ag + (long long)gr * K + kt + ac);
                    a0 = p[0];
                    a1 = p[1];
                    if (kt + 32 < K)
                        __builtin_prefetch((const void*)(Ag + (long long)gr * K + kt + 32 + ac), 0, 1);
                }
                *(uint4*)(&lA[0][ar * 32 + ac])     = a0;
                *(uint4*)(&lA[0][ar * 32 + ac + 8]) = a1;
            }
            {
                uint4 b0 = {0, 0, 0, 0};
                int gn = bn0 + bn;
                if (gn < N)
                    b0 = *(const uint4*)(Bt + (long long)gn * K + kt + bk);
                *(uint4*)(&lB[0][bn * 32 + bk]) = b0;
            }
            __syncthreads();
            compute(&lA[0][0], &lB[0][0]);
            __syncthreads();
        }
    }

    // ---- store: accum VGPR j -> row half*8+j, col = lane%16 ----
    for (int mi = 0; mi < 2; ++mi)
        for (int ni = 0; ni < 2; ++ni) {
            int col = bn0 + wn * 32 + ni * 16 + l16;
            if (col >= N) continue;
            float bv = bias ? bias[col] : 0.0f;
            for (int j = 0; j < 8; ++j) {
                int row = bm0 + wm * 32 + mi * 16 + half * 8 + j;
                if (row >= M) continue;
                long long idx = (long long)row * N + col;
                float v = acc[mi][ni][j] + bv;
                if (accumulate) v += Cg[idx];
                if (relu) v = fmaxf(v, 0.0f);
                Cg[idx] = v;
                if (Cbf) Cbf[idx] = f2bf(v);
            }
        }
}

// ---------------------- fused ReLU + residual + LayerNorm ------------------

__global__ void relu_res_ln_kernel(const float* __restrict__ conv_out,
                                   float* __restrict__ h,           // in: prev, out: new
                                   unsigned short* __restrict__ h_bf,
                                   const float* __restrict__ g,
                                   const float* __restrict__ b, int nrows) {
    int row  = blockIdx.x * 8 + (threadIdx.x >> 5);
    int lane = threadIdx.x & 31;
    if (row >= nrows) return;
    long long base = (long long)row * HD;
    float v[8];
    float s = 0.f;
    for (int i = 0; i < 8; ++i) {
        int c = lane + i * 32;
        float x = fmaxf(conv_out[base + c], 0.0f) + h[base + c];
        v[i] = x;
        s += x;
    }
    for (int off = 16; off; off >>= 1) s += __shfl_xor(s, off, 32);
    float mu = s * (1.0f / HD);
    float q = 0.f;
    for (int i = 0; i < 8; ++i) { float d = v[i] - mu; q += d * d; }
    for (int off = 16; off; off >>= 1) q += __shfl_xor(q, off, 32);
    float inv = rsqrtf(q * (1.0f / HD) + 1e-5f);
    for (int i = 0; i < 8; ++i) {
        int c = lane + i * 32;
        float y = (v[i] - mu) * inv * g[c] + b[c];
        h[base + c]    = y;
        h_bf[base + c] = f2bf(y);
    }
}

// ------------------------------ log_softmax --------------------------------

__global__ void log_softmax_kernel(float* __restrict__ out, int nrows, int c) {
    int row  = blockIdx.x * 8 + (threadIdx.x >> 5);
    int lane = threadIdx.x & 31;
    if (row >= nrows) return;
    float* p = out + (long long)row * c;
    float x0 = p[lane];
    float x1 = (lane + 32 < c) ? p[lane + 32] : -1e30f;
    float m = fmaxf(x0, x1);
    for (int off = 16; off; off >>= 1) m = fmaxf(m, __shfl_xor(m, off, 32));
    float s = __expf(x0 - m) + ((lane + 32 < c) ? __expf(x1 - m) : 0.0f);
    for (int off = 16; off; off >>= 1) s += __shfl_xor(s, off, 32);
    float lse = m + __logf(s);
    p[lane] = x0 - lse;
    if (lane + 32 < c) p[lane + 32] = x1 - lse;
}

// ------------------------------- launcher ----------------------------------

extern "C" void kernel_launch(void* const* d_in, const int* in_sizes, int n_in,
                              void* d_out, int out_size, void* d_ws, size_t ws_size,
                              hipStream_t stream) {
    (void)n_in; (void)out_size; (void)ws_size;

    const float* x_author     = (const float*)d_in[0];
    const float* x_paper      = (const float*)d_in[1];
    const int*   e0_src = (const int*)d_in[2];
    const int*   e0_dst = (const int*)d_in[3];
    const int*   e1_src = (const int*)d_in[4];
    const int*   e1_dst = (const int*)d_in[5];
    const int*   e2_src = (const int*)d_in[6];
    const int*   e2_dst = (const int*)d_in[7];
    const float* emb_W_author = (const float*)d_in[8];
    const float* emb_b_author = (const float*)d_in[9];
    const float* emb_W_paper  = (const float*)d_in[10];
    const float* emb_b_paper  = (const float*)d_in[11];
    const float* W_self  = (const float*)d_in[12];
    const float* W_neigh = (const float*)d_in[13];
    const float* b_conv  = (const float*)d_in[14];
    const float* ln_g_author = (const float*)d_in[15];
    const float* ln_b_author = (const float*)d_in[16];
    const float* ln_g_paper  = (const float*)d_in[17];
    const float* ln_b_paper  = (const float*)d_in[18];
    const float* pW1_author = (const float*)d_in[19];
    const float* pb1_author = (const float*)d_in[20];
    const float* pW2_author = (const float*)d_in[21];
    const float* pb2_author = (const float*)d_in[22];
    const float* pW1_paper  = (const float*)d_in[23];
    const float* pb1_paper  = (const float*)d_in[24];
    const float* pW2_paper  = (const float*)d_in[25];
    const int E = in_sizes[2];

    // ---- workspace carve-out (256B aligned) ----
    char* wp = (char*)d_ws;
    auto alloc = [&](size_t bytes) -> void* {
        void* p = (void*)wp;
        wp += (bytes + 255) & ~(size_t)255;
        return p;
    };
    float* ha    = (float*)alloc((size_t)NA * HD * 4);
    float* hp    = (float*)alloc((size_t)NP * HD * 4);
    float* out_a = (float*)alloc((size_t)NA * HD * 4);
    float* out_p = (float*)alloc((size_t)NP * HD * 4);
    float* agg   = (float*)alloc((size_t)NP * HD * 4);
    float* deg   = (float*)alloc((size_t)NP * 4);
    unsigned short* ha_bf  = (unsigned short*)alloc((size_t)NA * HD * 2);
    unsigned short* hp_bf  = (unsigned short*)alloc((size_t)NP * HD * 2);
    unsigned short* agg_bf = (unsigned short*)alloc((size_t)NP * HD * 2);
    unsigned short* xa_bf  = (unsigned short*)alloc((size_t)NA * DIN * 2);
    unsigned short* xp_bf  = (unsigned short*)alloc((size_t)NP * DIN * 2);
    unsigned short* embWa_bf = (unsigned short*)alloc((size_t)DIN * HD * 2);   // [HD][DIN]
    unsigned short* embWp_bf = (unsigned short*)alloc((size_t)DIN * HD * 2);
    unsigned short* Wself_bf = (unsigned short*)alloc((size_t)NLAYERS * 3 * HD * HD * 2); // [HD][HD] each
    unsigned short* Wnei_bf  = (unsigned short*)alloc((size_t)NLAYERS * 3 * HD * HD * 2);
    unsigned short* pW1a_bf  = (unsigned short*)alloc((size_t)HD * HD * 2);
    unsigned short* pW2a_bf  = (unsigned short*)alloc((size_t)HD * NC * 2);    // [NC][HD]
    unsigned short* pW1p_bf  = (unsigned short*)alloc((size_t)HD * HD * 2);
    unsigned short* pW2p_bf  = (unsigned short*)alloc((size_t)HD * HD * 2);

    float* oa = (float*)d_out;                 // [NA, NC]
    float* op = oa + (size_t)NA * NC;          // [NP, HD]

    auto cast = [&](const float* s, unsigned short* d, int n) {
        hipLaunchKernelGGL(cast_bf16_kernel, dim3((n + 255) / 256), dim3(256), 0, stream, s, d, n);
    };
    auto castT = [&](const float* s, unsigned short* d, int N_, int K_) {
        int n = N_ * K_;
        hipLaunchKernelGGL(cast_bf16_t_kernel, dim3((n + 255) / 256), dim3(256), 0, stream, s, d, N_, K_);
    };
    auto gemm = [&](const unsigned short* Ab, const unsigned short* Bt, const float* bias,
                    float* C, unsigned short* Cbf, int M, int N, int K, int acc, int relu) {
        dim3 grid((N + 63) / 64, (M + 127) / 128);
        hipLaunchKernelGGL(gemm_bf16_kernel, grid, dim3(256), 0, stream,
                           Ab, Bt, bias, C, Cbf, M, N, K, acc, relu);
    };
    auto zero = [&](float* p, long long n) {
        hipLaunchKernelGGL(fill_zero_kernel, dim3((unsigned)((n + 255) / 256)), dim3(256), 0, stream, p, n);
    };
    auto aggregate = [&](const float* hsrc, const int* src, const int* dst, int nDst) {
        zero(agg, (long long)nDst * HD);
        zero(deg, nDst);
        long long tot = (long long)E * (HD / 4);
        hipLaunchKernelGGL(scatter_mean_kernel, dim3((unsigned)((tot + 255) / 256)), dim3(256), 0,
                           stream, hsrc, src, dst, agg, deg, E);
        long long n = (long long)nDst * HD;
        hipLaunchKernelGGL(mean_cast_kernel, dim3((unsigned)((n + 255) / 256)), dim3(256), 0,
                           stream, agg, deg, agg_bf, n);
    };

    // ---- 1. cast activations; cast+transpose all weights ----
    cast(x_author, xa_bf, NA * DIN);
    cast(x_paper,  xp_bf, NP * DIN);
    castT(emb_W_author, embWa_bf, HD, DIN);
    castT(emb_W_paper,  embWp_bf, HD, DIN);
    for (int s = 0; s < NLAYERS * 3; ++s) {
        castT(W_self  + (size_t)s * HD * HD, Wself_bf + (size_t)s * HD * HD, HD, HD);
        castT(W_neigh + (size_t)s * HD * HD, Wnei_bf  + (size_t)s * HD * HD, HD, HD);
    }
    castT(pW1_author, pW1a_bf, HD, HD);
    castT(pW2_author, pW2a_bf, NC, HD);
    castT(pW1_paper,  pW1p_bf, HD, HD);
    castT(pW2_paper,  pW2p_bf, HD, HD);

    // ---- 2. input embeddings ----
    gemm(xa_bf, embWa_bf, emb_b_author, ha, ha_bf, NA, HD, DIN, 0, 0);
    gemm(xp_bf, embWp_bf, emb_b_paper,  hp, hp_bf, NP, HD, DIN, 0, 0);

    // ---- 3. SAGE layers ----
    for (int l = 0; l < NLAYERS; ++l) {
        const unsigned short* Ws = Wself_bf + (size_t)l * 3 * HD * HD;
        const unsigned short* Wn = Wnei_bf  + (size_t)l * 3 * HD * HD;
        const float* bc = b_conv + (size_t)l * 3 * HD;

        // edge type 0: author -> paper
        aggregate(ha, e0_src, e0_dst, NP);
        gemm(hp_bf,  Ws + 0 * HD * HD, bc + 0 * HD, out_p, nullptr, NP, HD, HD, 0, 0);
        gemm(agg_bf, Wn + 0 * HD * HD, nullptr,     out_p, nullptr, NP, HD, HD, 1, 0);
        // edge type 1: paper -> paper
        aggregate(hp, e1_src, e1_dst, NP);
        gemm(hp_bf,  Ws + 1 * HD * HD, bc + 1 * HD, out_p, nullptr, NP, HD, HD, 1, 0);
        gemm(agg_bf, Wn + 1 * HD * HD, nullptr,     out_p, nullptr, NP, HD, HD, 1, 0);
        // edge type 2: paper -> author
        aggregate(hp, e2_src, e2_dst, NA);
        gemm(ha_bf,  Ws + 2 * HD * HD, bc + 2 * HD, out_a, nullptr, NA, HD, HD, 0, 0);
        gemm(agg_bf, Wn + 2 * HD * HD, nullptr,     out_a, nullptr, NA, HD, HD, 1, 0);

        // fused relu + residual + LayerNorm (updates f32 + bf16 features)
        hipLaunchKernelGGL(relu_res_ln_kernel, dim3((NA + 7) / 8), dim3(256), 0, stream,
                           out_a, ha, ha_bf, ln_g_author, ln_b_author, NA);
        hipLaunchKernelGGL(relu_res_ln_kernel, dim3((NP + 7) / 8), dim3(256), 0, stream,
                           out_p, hp, hp_bf, ln_g_paper, ln_b_paper, NP);
    }

    // ---- 4. output heads ----
    gemm(ha_bf, pW1a_bf, pb1_author, out_a, agg_bf, NA, HD, HD, 0, 1);
    gemm(agg_bf, pW2a_bf, pb2_author, oa, nullptr, NA, NC, HD, 0, 0);
    hipLaunchKernelGGL(log_softmax_kernel, dim3((NA + 7) / 8), dim3(256), 0, stream, oa, NA, NC);
    gemm(hp_bf, pW1p_bf, pb1_paper, out_p, agg_bf, NP, HD, HD, 0, 1);
    gemm(agg_bf, pW2p_bf, nullptr, op, nullptr, NP, HD, HD, 0, 0);
}
